// QuantumGeometricAttention_23390391894111
// MI455X (gfx1250) — matrix operand order repaired and checked by reading, not compiled, not run on hardware
//
#include <hip/hip_runtime.h>
#include <hip/hip_bf16.h>

#define BB 2
#define NN 2048
#define DIMX 1024
#define HEADS 16
#define HD 64
#define BN (BB*NN)            // 4096 tokens
#define TOK (BB*HEADS*NN)     // 65536 head-tokens
#define QKVN 3072
#define PI_F 3.14159265358979323846f
#define SCALE_F 0.125f        // 64^-0.5
#define EPSN 1e-12f

typedef __attribute__((ext_vector_type(16))) __bf16 v16bf;
typedef __attribute__((ext_vector_type(8)))  __bf16 v8bf;
typedef __attribute__((ext_vector_type(8)))  float  v8f;
typedef __attribute__((ext_vector_type(4)))  unsigned int u32x4;
typedef __attribute__((ext_vector_type(8)))  int i32x8;
typedef __attribute__((ext_vector_type(4)))  int i32x4;

static __device__ __forceinline__ v8f wmma_bf16(v16bf a, v16bf b, v8f c) {
  // D = A(16x32 bf16) * B(32x16 bf16) + C(16x16 f32)
  return __builtin_amdgcn_wmma_f32_16x16x32_bf16(false, a, false, b, (short)0, c, false, false);
}

static __device__ __forceinline__ v16bf combine8(v8bf lo, v8bf hi) {
  return __builtin_shufflevector(lo, hi, 0,1,2,3,4,5,6,7,8,9,10,11,12,13,14,15);
}

// ---------------------------------------------------------------------------
// Tensor Data Mover: 2-D tile of 2-byte elements, global -> LDS (TENSORcnt).
// ---------------------------------------------------------------------------
static __device__ __forceinline__ void tdm_load_2d_bf16(unsigned lds_off, const void* gptr,
                                                        unsigned tile_w, unsigned tile_h,
                                                        unsigned tensor_w, unsigned tensor_h,
                                                        unsigned stride_elems) {
  unsigned long long ga = (unsigned long long)gptr;
  u32x4 g0;
  g0[0] = 1u;                                               // count=1, user mode
  g0[1] = lds_off;                                          // lds_addr (bytes)
  g0[2] = (unsigned)(ga & 0xFFFFFFFFu);                     // global_addr[31:0]
  g0[3] = (unsigned)((ga >> 32) & 0x1FFFFFFu) | (2u << 30); // global_addr[56:32], type=2
  i32x8 g1;
  g1[0] = (int)(1u << 16);                                  // wg_mask=0, data_size=1 (2B)
  g1[1] = (int)((tensor_w & 0xFFFFu) << 16);                // tensor_dim0[15:0]
  g1[2] = (int)((tensor_w >> 16) | ((tensor_h & 0xFFFFu) << 16)); // dim0 hi | dim1 lo
  g1[3] = (int)((tensor_h >> 16) | (tile_w << 16));         // dim1 hi | tile_dim0
  g1[4] = (int)(tile_h & 0xFFFFu);                          // tile_dim1 (tile_dim2=0)
  g1[5] = (int)stride_elems;                                // tensor_dim0_stride lo
  g1[6] = 0;
  g1[7] = 0;
  i32x4 gz4 = {0, 0, 0, 0};
  i32x8 gz8 = {0, 0, 0, 0, 0, 0, 0, 0};
  __builtin_amdgcn_tensor_load_to_lds(g0, g1, gz4, gz4, gz8, 0);
}

// ---------------------------------------------------------------------------
// K0a: x fp32 -> bf16 ;  K0b: Wqkv fp32 -> bf16 TRANSPOSED (WbT[n][k])
// ---------------------------------------------------------------------------
__global__ void k0_tobf16(const float* __restrict__ src, __bf16* __restrict__ dst, int n) {
  int i = blockIdx.x * 256 + threadIdx.x;
  if (i < n) dst[i] = (__bf16)src[i];
}
__global__ void k0_twbf16(const float* __restrict__ W, __bf16* __restrict__ WbT) {
  int i = blockIdx.x * 256 + threadIdx.x;        // over 3072*1024
  if (i < QKVN * DIMX) {
    int n = i >> 10, k = i & (DIMX - 1);
    WbT[i] = (__bf16)W[k * QKVN + n];
  }
}

// ---------------------------------------------------------------------------
// K1: qkv_raw = x @ Wqkv + bqkv.  bf16 WMMA; TDM double-buffered W tiles
// staged K-major (from WbT) so B-fragments are two 16-byte LDS reads.
// ---------------------------------------------------------------------------
__global__ void k1_qkv_gemm(const __bf16* __restrict__ Xb, const __bf16* __restrict__ WbT,
                            const float* __restrict__ bias, float* __restrict__ out) {
  __shared__ __bf16 bs[2][32][32] __attribute__((aligned(64)));   // [buf][n][k]
  const int tid  = threadIdx.x;
  const int lane = tid & 31;
  const int wv   = tid >> 5;
  const int half = lane >> 4;
  const int l15  = lane & 15;
  const int mbase = blockIdx.y * 32 + (wv >> 1) * 16;
  const int nblk  = blockIdx.x * 32;
  const int nloc  = (wv & 1) * 16 + l15;
  const int m = mbase + l15;
  const unsigned bs_off0 = (unsigned)(uintptr_t)&bs[0][0][0];
  const unsigned bs_off1 = (unsigned)(uintptr_t)&bs[1][0][0];
  const __bf16* wtile = WbT + (size_t)nblk * DIMX;

  if (wv == 0) tdm_load_2d_bf16(bs_off0, wtile, 32, 32, 32, 32, DIMX);

  v8f acc = {0.f,0.f,0.f,0.f,0.f,0.f,0.f,0.f};
  for (int ks = 0; ks < DIMX / 32; ++ks) {
    __syncthreads();                     // prior reads of other buffer done
    if (wv == 0) {
      __builtin_amdgcn_s_wait_tensorcnt(0);          // tile ks landed
      if (ks + 1 < DIMX / 32)
        tdm_load_2d_bf16((ks & 1) ? bs_off0 : bs_off1,
                         wtile + (ks + 1) * 32, 32, 32, 32, 32, DIMX);
    }
    __syncthreads();                     // tile ks visible to all waves
    const int kk = ks * 32;
    const __bf16* xp = Xb + m * DIMX + kk + half * 8;
    v16bf a = combine8(*(const v8bf*)xp, *(const v8bf*)(xp + 16));
    v16bf b = combine8(*(const v8bf*)&bs[ks & 1][nloc][half * 8], *(const v8bf*)&bs[ks & 1][nloc][half * 8 + 16]);
    acc = wmma_bf16(a, b, acc);
  }
  const int ncol = nblk + nloc;
  const float bia = bias[ncol];
#pragma unroll
  for (int r = 0; r < 8; ++r) {
    int row = mbase + r + half * 8;
    out[row * QKVN + ncol] = acc[r] + bia;
  }
}

// ---------------------------------------------------------------------------
// K2: in-place l2norm over each 3072-wide qkv row
// ---------------------------------------------------------------------------
__global__ void k2_rownorm(float* __restrict__ qkv) {
  __shared__ float red[256];
  const int t = threadIdx.x;
  float* row = qkv + (size_t)blockIdx.x * QKVN;
  float s = 0.f;
  for (int i = t; i < QKVN; i += 256) { float v = row[i]; s += v * v; }
  red[t] = s; __syncthreads();
  for (int off = 128; off > 0; off >>= 1) {
    if (t < off) red[t] += red[t + off];
    __syncthreads();
  }
  const float inv = 1.f / fmaxf(sqrtf(red[0]), EPSN);
  for (int i = t; i < QKVN; i += 256) row[i] *= inv;
}

// ---------------------------------------------------------------------------
// K3: prepare quantum state per head-token.
// ---------------------------------------------------------------------------
__global__ void k3_prep(const float* __restrict__ qkvn,
                        const float* __restrict__ Wp1, const float* __restrict__ bp1,
                        const float* __restrict__ Wp2, const float* __restrict__ bp2,
                        float* __restrict__ qre, float* __restrict__ qim,
                        float* __restrict__ kre, float* __restrict__ kim,
                        __bf16* __restrict__ vtb) {
  __shared__ float tsh[64];
  __shared__ float ph1[128];
  __shared__ float red[64];
  const int gid = blockIdx.x;
  const int sel = gid >> 16;       // 0=q 1=k 2=v
  const int idx = gid & 0xFFFF;    // (b*16+h)*2048+n
  const int b = idx >> 15;
  const int h = (idx >> 11) & 15;
  const int n = idx & 2047;
  const int t = threadIdx.x;
  const float* src = qkvn + (size_t)(b * NN + n) * QKVN + sel * DIMX + h * HD;
  if (t < 64) tsh[t] = src[t];
  __syncthreads();
  if (t < 64) red[t] = tsh[t] * tsh[t];
  __syncthreads();
  for (int off = 32; off > 0; off >>= 1) {
    if (t < off) red[t] += red[t + off];
    __syncthreads();
  }
  const float inv = 1.f / fmaxf(sqrtf(red[0]), EPSN);
  if (sel == 2) {
    if (t < 64) {
      const int bh = idx >> 11;
      vtb[((size_t)bh * HD + t) * NN + n] = (__bf16)(fabsf(tsh[t]) * inv);
    }
    return;
  }
  float a = bp1[t];
  for (int d = 0; d < 64; ++d) a += tsh[d] * Wp1[d * 128 + t];
  ph1[t] = tanhf(a);
  __syncthreads();
  if (t < 64) {
    float p = bp2[t];
    for (int j = 0; j < 128; ++j) p += ph1[j] * Wp2[j * 64 + t];
    p = tanhf(p) * PI_F;
    float u = tsh[t] * inv;
    float sn, cs; __sincosf(p, &sn, &cs);
    float* ore = (sel == 0) ? qre : kre;
    float* oim = (sel == 0) ? qim : kim;
    ore[(size_t)idx * HD + t] = u * cs;
    oim[(size_t)idx * HD + t] = u * sn;
  }
}

// ---------------------------------------------------------------------------
// K4a: M = (I + 0.1*Wf)^10
// ---------------------------------------------------------------------------
__global__ void k4a_matpow(const float* __restrict__ Wf, float* __restrict__ M) {
  __shared__ float A[4096];
  __shared__ float Mc[4096];
  __shared__ float T[4096];
  const int t = threadIdx.x;
  for (int i = t; i < 4096; i += 256) {
    int r = i >> 6, c = i & 63;
    float v = 0.1f * Wf[i] + ((r == c) ? 1.f : 0.f);
    A[i] = v; Mc[i] = v;
  }
  __syncthreads();
  for (int s = 0; s < 9; ++s) {
    for (int i = t; i < 4096; i += 256) {
      int r = i >> 6, c = i & 63;
      float acc = 0.f;
      for (int k = 0; k < 64; ++k) acc += Mc[r * 64 + k] * A[k * 64 + c];
      T[i] = acc;
    }
    __syncthreads();
    for (int i = t; i < 4096; i += 256) Mc[i] = T[i];
    __syncthreads();
  }
  for (int i = t; i < 4096; i += 256) M[i] = Mc[i];
}

// ---------------------------------------------------------------------------
// K4b: z = l2norm(z @ M), complex z. In place.
// ---------------------------------------------------------------------------
__global__ void k4b_flow(float* __restrict__ qre, float* __restrict__ qim,
                         float* __restrict__ kre, float* __restrict__ kim,
                         const float* __restrict__ M) {
  __shared__ float zr[64], zi[64], red[64];
  const int tensor = blockIdx.x >> 16;
  const int tok    = blockIdx.x & 0xFFFF;
  float* re = tensor ? kre : qre;
  float* im = tensor ? kim : qim;
  const int t = threadIdx.x;
  zr[t] = re[(size_t)tok * HD + t];
  zi[t] = im[(size_t)tok * HD + t];
  __syncthreads();
  float orr = 0.f, oii = 0.f;
  for (int d = 0; d < 64; ++d) {
    float mm = M[d * 64 + t];
    orr += zr[d] * mm;
    oii += zi[d] * mm;
  }
  red[t] = orr * orr + oii * oii;
  __syncthreads();
  for (int off = 32; off > 0; off >>= 1) {
    if (t < off) red[t] += red[t + off];
    __syncthreads();
  }
  const float inv = 1.f / fmaxf(sqrtf(red[0]), EPSN);
  re[(size_t)tok * HD + t] = orr * inv;
  im[(size_t)tok * HD + t] = oii * inv;
}

// ---------------------------------------------------------------------------
// K5: parallel transport; emits normalized k_t as bf16 (kbr/kbi).
// ---------------------------------------------------------------------------
__global__ void k5_transport(const float* __restrict__ qre, const float* __restrict__ qim,
                             const float* __restrict__ kre, const float* __restrict__ kim,
                             __bf16* __restrict__ kbr, __bf16* __restrict__ kbi) {
  __shared__ float rr[64], ri[64];
  __shared__ float tfs[2];
  const int t = threadIdx.x;
  const size_t base = (size_t)blockIdx.x * HD;
  const float qr = qre[base + t], qi = qim[base + t];
  const float kr = kre[base + t], ki = kim[base + t];
  const float ar = kr - qr, ai = ki - qi;
  rr[t] = ar * ar - ai * ai;
  ri[t] = 2.f * ar * ai;
  __syncthreads();
  for (int off = 32; off > 0; off >>= 1) {
    if (t < off) { rr[t] += rr[t + off]; ri[t] += ri[t + off]; }
    __syncthreads();
  }
  if (t == 0) {
    float sr = rr[0], si = ri[0];
    float rad = sqrtf(sr * sr + si * si);
    float dre  = sqrtf(fmaxf(0.5f * (rad + sr), 0.f));
    float dimm = copysignf(sqrtf(fmaxf(0.5f * (rad - sr), 0.f)), si);
    float zrr = 10.f * dre, zii = 10.f * dimm;
    float tfr, tfi;
    if (zrr > 80.f) { tfr = 0.f; tfi = 0.f; }
    else {
      float ex = __expf(zrr);
      float wr = ex * __cosf(zii), wi = ex * __sinf(zii);
      float denr = 1.f + wr, deni = wi;
      float d2 = fmaxf(denr * denr + deni * deni, 1e-30f);
      tfr = denr / d2; tfi = -deni / d2;
    }
    tfs[0] = tfr; tfs[1] = tfi;
  }
  __syncthreads();
  const float tfr = tfs[0], tfi = tfs[1];
  const float omr = 1.f - tfr, omi = -tfi;
  const float ktr = kr * tfr - ki * tfi + qr * omr - qi * omi;
  const float kti = kr * tfi + ki * tfr + qr * omi + qi * omr;
  rr[t] = ktr * ktr + kti * kti;
  __syncthreads();
  for (int off = 32; off > 0; off >>= 1) {
    if (t < off) rr[t] += rr[t + off];
    __syncthreads();
  }
  const float inv = 1.f / fmaxf(sqrtf(rr[0]), EPSN);
  kbr[base + t] = (__bf16)(ktr * inv);
  kbi[base + t] = (__bf16)(kti * inv);
}

// ---------------------------------------------------------------------------
// K6: flash attention, bf16 WMMA; double-buffered TDM staging overlapped
// with compute. Scores computed transposed so S^T's C-layout is the
// A-fragment for P@V. All LDS fragment reads are 16/32-byte vector loads.
// ---------------------------------------------------------------------------
__global__ void k6_attn(const float* __restrict__ qre, const float* __restrict__ qim,
                        const __bf16* __restrict__ kbr, const __bf16* __restrict__ kbi,
                        const __bf16* __restrict__ vtb, float* __restrict__ out) {
  __shared__ __bf16 kr_s[2][32][64] __attribute__((aligned(64)));
  __shared__ __bf16 ki_s[2][32][64] __attribute__((aligned(64)));
  __shared__ __bf16 vT_s[2][64][32] __attribute__((aligned(64)));
  const int tid  = threadIdx.x;
  const int lane = tid & 31;
  const int wv   = tid >> 5;
  const int half = lane >> 4;
  const int l15  = lane & 15;
  const int bh = blockIdx.y;
  const int n0 = blockIdx.x * 64 + wv * 16;
  const size_t tq = (size_t)bh * NN + n0;
  unsigned kroff[2] = {(unsigned)(uintptr_t)&kr_s[0][0][0], (unsigned)(uintptr_t)&kr_s[1][0][0]};
  unsigned kioff[2] = {(unsigned)(uintptr_t)&ki_s[0][0][0], (unsigned)(uintptr_t)&ki_s[1][0][0]};
  unsigned vtoff[2] = {(unsigned)(uintptr_t)&vT_s[0][0][0], (unsigned)(uintptr_t)&vT_s[1][0][0]};

  auto issue_tile = [&](int kb, int buf) {
    const size_t tk = ((size_t)bh * NN + kb * 32) * HD;     // element offset
    tdm_load_2d_bf16(kroff[buf], kbr + tk, 2048, 1, 2048, 1, 2048);
    tdm_load_2d_bf16(kioff[buf], kbi + tk, 2048, 1, 2048, 1, 2048);
    tdm_load_2d_bf16(vtoff[buf], vtb + (size_t)bh * HD * NN + kb * 32, 32, 64, 32, 64, NN);
  };

  // per-wave Q B-fragments (imag pre-negated: S = Qr.Ktr^T - Qi.Kti^T)
  v16bf qrB[2], qiB[2];
  {
    const float* qr = qre + (tq + l15) * HD;
    const float* qi = qim + (tq + l15) * HD;
#pragma unroll
    for (int ks2 = 0; ks2 < 2; ++ks2) {
      int d0 = ks2 * 32 + half * 8;
#pragma unroll
      for (int e = 0; e < 16; ++e) {
        qrB[ks2][e] = (__bf16)qr[d0 + e + (e >= 8 ? 8 : 0)];
        qiB[ks2][e] = (__bf16)(-qi[d0 + e + (e >= 8 ? 8 : 0)]);
      }
    }
  }

  if (wv == 0) issue_tile(0, 0);

  const v8f vzero = {0.f,0.f,0.f,0.f,0.f,0.f,0.f,0.f};
  v8f o[4] = {vzero, vzero, vzero, vzero};
  float mrow = -__builtin_inff();
  float lsum = 0.f;

  for (int kb = 0; kb < NN / 32; ++kb) {
    __syncthreads();                     // reads of the other buffer finished
    if (wv == 0) {
      __builtin_amdgcn_s_wait_tensorcnt(0);          // tile kb landed
      if (kb + 1 < NN / 32) issue_tile(kb + 1, (kb + 1) & 1);
    }
    __syncthreads();                     // tile kb visible; kb+1 DMA in flight
    const int buf = kb & 1;

    // S^T tiles: st[tkey] holds S^T[key = tkey*16 + r + 8*half][q = l15]
    v8f st[2];
#pragma unroll
    for (int tkey = 0; tkey < 2; ++tkey) {
      const int key = tkey * 16 + l15;
      v8f acc = vzero;
#pragma unroll
      for (int ks2 = 0; ks2 < 2; ++ks2) {
        const int d0 = ks2 * 32 + half * 8;
        v16bf arf = combine8(*(const v8bf*)&kr_s[buf][key][d0],
                             *(const v8bf*)&kr_s[buf][key][d0 + 16]);
        v16bf aif = combine8(*(const v8bf*)&ki_s[buf][key][d0],
                             *(const v8bf*)&ki_s[buf][key][d0 + 16]);
        acc = wmma_bf16(arf, qrB[ks2], acc);
        acc = wmma_bf16(aif, qiB[ks2], acc);
      }
      st[tkey] = acc;
    }

    // online softmax; lane owns query l15 (duplicated across wave halves)
    float p0[8], p1[8];
    float tmax = -__builtin_inff();
#pragma unroll
    for (int r = 0; r < 8; ++r) {
      p0[r] = st[0][r] * SCALE_F;
      p1[r] = st[1][r] * SCALE_F;
      tmax = fmaxf(tmax, fmaxf(p0[r], p1[r]));
    }
    tmax = fmaxf(tmax, __shfl_xor(tmax, 16, 32));
    const float mn = fmaxf(mrow, tmax);
    const float alpha = __expf(mrow - mn);
    float tsum = 0.f;
#pragma unroll
    for (int r = 0; r < 8; ++r) {
      p0[r] = __expf(p0[r] - mn);
      p1[r] = __expf(p1[r] - mn);
      tsum += p0[r] + p1[r];
    }
    tsum += __shfl_xor(tsum, 16, 32);
    lsum = lsum * alpha + tsum;
    mrow = mn;

#pragma unroll
    for (int r = 0; r < 8; ++r) {
      const int row = r + half * 8;
      const float arow = __shfl(alpha, (lane & 16) | row, 32);
#pragma unroll
      for (int f = 0; f < 4; ++f) o[f][r] *= arow;
    }

    // P as A-fragment: elem e<8 -> key e+8*half (st[0]); e>=8 -> +16 (st[1])
    v16bf pa;
#pragma unroll
    for (int r = 0; r < 8; ++r) { pa[r] = (__bf16)p0[r]; pa[8 + r] = (__bf16)p1[r]; }

#pragma unroll
    for (int f = 0; f < 4; ++f) {
      const int d = f * 16 + l15;
      v16bf vb = combine8(*(const v8bf*)&vT_s[buf][d][half * 8], *(const v8bf*)&vT_s[buf][d][half * 8 + 16]);
      o[f] = wmma_bf16(pa, vb, o[f]);
    }
  }

  const int b = bh >> 4, h = bh & 15;
#pragma unroll
  for (int r = 0; r < 8; ++r) {
    const int row = r + half * 8;
    const float lr = __shfl(lsum, (lane & 16) | row, 32);
    const float linv = 1.f / lr;
    const int n = n0 + row;
#pragma unroll
    for (int f = 0; f < 4; ++f) {
      const int d = f * 16 + l15;
      out[(((size_t)b * NN + n) * HEADS + h) * HD + d] = o[f][r] * linv;
    }
  }
}

// ---------------------------------------------------------------------------
extern "C" void kernel_launch(void* const* d_in, const int* in_sizes, int n_in,
                              void* d_out, int out_size, void* d_ws, size_t ws_size,
                              hipStream_t stream) {
  const float* x    = (const float*)d_in[0];
  const float* Wqkv = (const float*)d_in[1];
  const float* bqkv = (const float*)d_in[2];
  const float* Wp1  = (const float*)d_in[3];
  const float* bp1  = (const float*)d_in[4];
  const float* Wp2  = (const float*)d_in[5];
  const float* bp2  = (const float*)d_in[6];
  const float* Wf   = (const float*)d_in[7];
  float* out = (float*)d_out;

  char* p = (char*)d_ws;
  auto alloc = [&](size_t bytes) { char* r = p; p += (bytes + 255) & ~(size_t)255; return r; };
  float*  qkvn = (float*) alloc((size_t)BN * QKVN * 4);
  float*  qre  = (float*) alloc((size_t)TOK * HD * 4);
  float*  qim  = (float*) alloc((size_t)TOK * HD * 4);
  float*  kre  = (float*) alloc((size_t)TOK * HD * 4);
  float*  kim  = (float*) alloc((size_t)TOK * HD * 4);
  float*  M    = (float*) alloc(4096 * 4);
  __bf16* Xb   = (__bf16*)alloc((size_t)BN * DIMX * 2);
  __bf16* WbT  = (__bf16*)alloc((size_t)DIMX * QKVN * 2);
  __bf16* kbr  = (__bf16*)alloc((size_t)TOK * HD * 2);
  __bf16* kbi  = (__bf16*)alloc((size_t)TOK * HD * 2);
  __bf16* vtb  = (__bf16*)alloc((size_t)TOK * HD * 2);

  const int nX = BN * DIMX, nW = DIMX * QKVN;
  k0_tobf16<<<dim3((nX + 255) / 256), 256, 0, stream>>>(x, Xb, nX);
  k0_twbf16<<<dim3((nW + 255) / 256), 256, 0, stream>>>(Wqkv, WbT);
  k1_qkv_gemm<<<dim3(QKVN / 32, BN / 32), 128, 0, stream>>>(Xb, WbT, bqkv, qkvn);
  k2_rownorm<<<dim3(BN), 256, 0, stream>>>(qkvn);
  k3_prep<<<dim3(3 * TOK), 128, 0, stream>>>(qkvn, Wp1, bp1, Wp2, bp2,
                                             qre, qim, kre, kim, vtb);
  k4a_matpow<<<dim3(1), 256, 0, stream>>>(Wf, M);
  k4b_flow<<<dim3(2 * TOK), 64, 0, stream>>>(qre, qim, kre, kim, M);
  k5_transport<<<dim3(TOK), 64, 0, stream>>>(qre, qim, kre, kim, kbr, kbi);
  k6_attn<<<dim3(NN / 64, BB * HEADS), 128, 0, stream>>>(qre, qim, kbr, kbi, vtb, out);
}
